// ContextGNNNetwork_87531433493175
// MI455X (gfx1250) — matrix operand order repaired and checked by reading, not compiled, not run on hardware
//
#include <hip/hip_runtime.h>
#include <hip/hip_bf16.h>

typedef __attribute__((ext_vector_type(16))) _Float16 v16h;
typedef __attribute__((ext_vector_type(8)))  float    v8f;

#define HID   256
#define BATCH 1024
#define NACT  1024

// ---------------------------------------------------------------------------
// C[M,N] = act( A[M,K] @ W[K,N] + bias )   (f32 in/out, f16 WMMA core)
// N, K compile-time: store offsets fold into global_store immediates, and
// A-row addressing strength-reduces to shifts.
// grid = (N/32, M/128), block = 256 threads = 8 wave32s.
// Each wave computes a 16(M) x 32(N) tile: one A fragment feeds two WMMAs.
// A fragment (16x32 f16): lanes 0..15 -> M=lane, lanes 16..31 -> M=lane-16,
//   element j -> k = (j&7) + 16*(j>>3) + 8*(lane>=16)
// B fragment (32x16 f16): N = lane&15, element j -> same k as A
// C/D (16x16 f32, 8 VGPRs): reg r, lane -> M = r + 8*(lane>=16), N = lane&15
// ---------------------------------------------------------------------------
template<int N, int K, bool RELU, bool HAS_BIAS>
__global__ __launch_bounds__(256)
void gemm_wmma_f16(const float* __restrict__ A, const float* __restrict__ W,
                   const float* __restrict__ bias, float* __restrict__ C)
{
    const int lane = threadIdx.x & 31;
    const int wave = threadIdx.x >> 5;
    const int hi   = lane >> 4;          // which lane-half
    const int mrow = lane & 15;
    const int n0   = blockIdx.x * 32;
    const int m0   = blockIdx.y * 128 + wave * 16;

    const float* __restrict__ Arow = A + (size_t)(m0 + mrow) * K;
    const float* __restrict__ Wrow = W + n0 + mrow;  // N-col = n0 + (lane & 15)

    v8f acc0 = {}, acc1 = {};
    #pragma unroll 2
    for (int k0 = 0; k0 < K; k0 += 32) {
        v16h a, b0, b1;
        #pragma unroll
        for (int j = 0; j < 16; ++j) {
            const int k = k0 + (j & 7) + ((j >> 3) << 4) + (hi << 3);
            a[j] = (_Float16)Arow[k];
        }
        const float* __restrict__ wp = Wrow + (size_t)k0 * N;
        #pragma unroll
        for (int j = 0; j < 16; ++j) {
            const int kk = (j & 7) + ((j >> 3) << 4) + (hi << 3);
            b0[j] = (_Float16)wp[(size_t)kk * N];
            b1[j] = (_Float16)wp[(size_t)kk * N + 16];
        }
        acc0 = __builtin_amdgcn_wmma_f32_16x16x32_f16(false, a, false, b0,
                                                      (short)0, acc0, false, false);
        acc1 = __builtin_amdgcn_wmma_f32_16x16x32_f16(false, a, false, b1,
                                                      (short)0, acc1, false, false);
    }

    const int n = n0 + mrow;
    float bias0 = 0.0f, bias1 = 0.0f;
    if (HAS_BIAS) { bias0 = bias[n]; bias1 = bias[n + 16]; }

    // Single base pointer; r*N is now a compile-time constant -> one s_clause
    // of immediate-offset global_store_b32.
    float* __restrict__ cp = C + (size_t)(m0 + (hi << 3)) * N + n;
    #pragma unroll
    for (int r = 0; r < 8; ++r) {
        float v0 = acc0[r] + bias0;
        float v1 = acc1[r] + bias1;
        if (RELU) { v0 = fmaxf(v0, 0.0f); v1 = fmaxf(v1, 0.0f); }
        cp[r * N]      = v0;
        cp[r * N + 16] = v1;
    }
}

// ---------------------------------------------------------------------------
// scores[b,a] = sum_h relu(U[b,h] + V[a,h]) * w2[h] + bs2
// grid = (NACT/32, BATCH/32), block = 256.
// LDS tiles stored [h][row] with stride 33 (odd) -> conflict-free writes
// (stride-33 walk covers all 64 banks) and conflict-free compute reads
// (V: lane-consecutive banks; U: wave-uniform broadcast).
// ---------------------------------------------------------------------------
#define HC 128   // h-chunk per staging pass (2 passes over HID=256)

__global__ __launch_bounds__(256)
void pairwise_score(const float* __restrict__ U, const float* __restrict__ V,
                    const float* __restrict__ w2, const float* __restrict__ bs2,
                    float* __restrict__ out)
{
    __shared__ float Us[HC * 33];
    __shared__ float Vs[HC * 33];
    __shared__ float ws[HC];

    const int t  = threadIdx.x;
    const int a0 = blockIdx.x * 32;
    const int b0 = blockIdx.y * 32;
    const int tx = t & 31;       // a-index within tile (== lane)
    const int wv = t >> 5;       // wave id 0..7 -> b rows {wv, wv+8, wv+16, wv+24}

    float acc[4] = {0.f, 0.f, 0.f, 0.f};

    for (int h0 = 0; h0 < HID; h0 += HC) {
        // Stage 32 x HC tiles, transposed into [h][row]
        #pragma unroll
        for (int i = 0; i < (32 * HC) / 256; ++i) {
            const int idx = t + 256 * i;
            const int h   = idx & (HC - 1);
            const int r   = idx >> 7;            // 0..31 (HC == 128)
            Us[h * 33 + r] = U[(size_t)(b0 + r) * HID + h0 + h];
            Vs[h * 33 + r] = V[(size_t)(a0 + r) * HID + h0 + h];
        }
        if (t < HC) ws[t] = w2[h0 + t];

        // Pull next h-chunk toward this WGP while we compute on this one
        // (lowers to global_prefetch_b8).
        if (h0 == 0) {
            const int r = t >> 3;                        // 0..31
            const int hq = (t & 7) * 16;                 // 0,16,...,112
            __builtin_prefetch(&U[(size_t)(b0 + r) * HID + HC + hq], 0, 1);
            __builtin_prefetch(&V[(size_t)(a0 + r) * HID + HC + hq], 0, 1);
        }
        __syncthreads();

        #pragma unroll 4
        for (int h = 0; h < HC; ++h) {
            const float v  = Vs[h * 33 + tx];    // per-lane, conflict-free
            const float wh = ws[h];              // broadcast
            #pragma unroll
            for (int q = 0; q < 4; ++q) {
                const float u = Us[h * 33 + wv + 8 * q];   // wave-uniform broadcast
                const float s = fmaxf(u + v, 0.0f);
                acc[q] = fmaf(s, wh, acc[q]);
            }
        }
        __syncthreads();
    }

    const float bias = bs2[0];
    #pragma unroll
    for (int q = 0; q < 4; ++q) {
        const int b = b0 + wv + 8 * q;
        out[(size_t)b * NACT + a0 + tx] = acc[q] + bias;
    }
}

// ---------------------------------------------------------------------------
extern "C" void kernel_launch(void* const* d_in, const int* in_sizes, int n_in,
                              void* d_out, int out_size, void* d_ws, size_t ws_size,
                              hipStream_t stream)
{
    const float* context = (const float*)d_in[0];   // 1024 x 512
    const float* action  = (const float*)d_in[1];   // 1024 x 256
    const float* Wc1 = (const float*)d_in[2];       // 512 x 256
    const float* bc1 = (const float*)d_in[3];
    const float* Wc2 = (const float*)d_in[4];       // 256 x 256
    const float* bc2 = (const float*)d_in[5];
    const float* Wa1 = (const float*)d_in[6];       // 256 x 256
    const float* ba1 = (const float*)d_in[7];
    const float* Wa2 = (const float*)d_in[8];       // 256 x 256
    const float* ba2 = (const float*)d_in[9];
    const float* Ws1 = (const float*)d_in[10];      // 512 x 256 (top = _c, bottom = _a)
    const float* bs1 = (const float*)d_in[11];
    const float* Ws2 = (const float*)d_in[12];      // 256 x 1
    const float* bs2 = (const float*)d_in[13];
    float* out = (float*)d_out;                     // 1024 x 1024

    // Workspace: 4 x (1024*256) f32 buffers, with liveness-based reuse.
    const size_t SZ = (size_t)BATCH * HID;
    float* buf0 = (float*)d_ws;          // hc1, then U
    float* buf1 = buf0 + SZ;             // ce
    float* buf2 = buf1 + SZ;             // ha1, then V
    float* buf3 = buf2 + SZ;             // ae

    const dim3 blk(256);
    const dim3 ggrid(HID / 32, BATCH / 128);   // (8, 8)

    // Context encoder
    gemm_wmma_f16<HID, 512, true,  true ><<<ggrid, blk, 0, stream>>>(context, Wc1, bc1, buf0);
    gemm_wmma_f16<HID, 256, true,  true ><<<ggrid, blk, 0, stream>>>(buf0,    Wc2, bc2, buf1);
    // U = ce @ Ws1_c   (hc1 dead -> reuse buf0)
    gemm_wmma_f16<HID, 256, false, false><<<ggrid, blk, 0, stream>>>(buf1, Ws1, nullptr, buf0);
    // Action encoder
    gemm_wmma_f16<HID, 256, true,  true ><<<ggrid, blk, 0, stream>>>(action, Wa1, ba1, buf2);
    gemm_wmma_f16<HID, 256, true,  true ><<<ggrid, blk, 0, stream>>>(buf2,   Wa2, ba2, buf3);
    // V = ae @ Ws1_a + bs1   (ha1 dead -> reuse buf2)
    gemm_wmma_f16<HID, 256, false, true ><<<ggrid, blk, 0, stream>>>(buf3, Ws1 + 256 * 256, bs1, buf2);

    // Pairwise relu-interaction + reduction (not expressible as GEMM)
    pairwise_score<<<dim3(NACT / 32, BATCH / 32), blk, 0, stream>>>(buf0, buf2, Ws2, bs2, out);
}
